// GCN_72327249265076
// MI455X (gfx1250) — compile-verified, matrix operand order checked by reading, not run on hardware
//
#include <hip/hip_runtime.h>
#include <hip/hip_bf16.h>

// ---------------------------------------------------------------------------
// GCN forward for MI455X (gfx1250, wave32, WMMA).
// Heavy GEMMs: bf16 hi/lo split (3x v_wmma_f32_16x16x32_bf16) for ~fp32
// accuracy. GEMM1->GEMM2 fused through LDS (support never touches HBM).
// B operands pre-split, pair-packed, and laid out so every inner-loop load
// is a b128 at a constant immediate offset from one running base pointer.
// ---------------------------------------------------------------------------

typedef __attribute__((ext_vector_type(16))) __bf16 v16bf;
typedef __attribute__((ext_vector_type(8)))  float  v8f;

#define B_SZ    1024
#define NN      264            // graph nodes
#define HID     256
#define FLAT    16896          // 132 * 128
#define W1LD    144            // K pairs per W1 column (288/2)
#define W1HL    (HID * W1LD)   // dword offset hi -> lo  (147,456 B)
#define SUPLD   148            // K pairs per support column in LDS (16B-aligned, bank-friendly)
#define FC1LD   8448           // K pairs per fc1 column (16896/2)
#define FC1HL   (128 * FC1LD)  // dword offset hi -> lo  (4,325,376 B, fits 24-bit imm)

union BF16Frag { v16bf v; unsigned short u[16]; unsigned int w[8]; };
union AccFrag  { v8f v; float f[8]; };

__device__ __forceinline__ unsigned short f2bf(float f) {
  unsigned int u = __float_as_uint(f);
  u += 0x7FFFu + ((u >> 16) & 1u);          // round-to-nearest-even
  return (unsigned short)(u >> 16);
}
__device__ __forceinline__ float bf2f(unsigned short s) {
  return __uint_as_float(((unsigned int)s) << 16);
}
__device__ __forceinline__ void split2(float x, unsigned short &h, unsigned short &l) {
  h = f2bf(x);
  l = f2bf(x - bf2f(h));
}
__device__ __forceinline__ unsigned int pack2(unsigned short lo16, unsigned short hi16) {
  return (unsigned int)lo16 | ((unsigned int)hi16 << 16);
}
__device__ __forceinline__ float selu_f(float x) {
  const float a = 1.6732632423543772f, s = 1.0507009873554805f;
  return x > 0.f ? s * x : s * a * (__expf(x) - 1.f);
}
__device__ __forceinline__ void frag_from_u4(BF16Frag &f, uint4 a, uint4 b) {
  f.w[0] = a.x; f.w[1] = a.y; f.w[2] = a.z; f.w[3] = a.w;
  f.w[4] = b.x; f.w[5] = b.y; f.w[6] = b.z; f.w[7] = b.w;
}
// A fragment from an f32 row: 8 pairs at fixed offsets {0,2,4,6,16,18,20,22}
// from (row + 8*half), split once into hi/lo bf16.
__device__ __forceinline__ void load_split_A(const float* __restrict__ arun,
                                             BF16Frag &ah, BF16Frag &al) {
  #pragma unroll
  for (int j = 0; j < 8; ++j) {
    const int o = ((j & 4) << 2) + ((j & 3) << 1);
    const float2 a = *(const float2*)(arun + o);
    split2(a.x, ah.u[2*j],   al.u[2*j]);
    split2(a.y, ah.u[2*j+1], al.u[2*j+1]);
  }
}

#define WMMA3(ACC, AH, AL, BH, BL)                                                             \
  (ACC).v = __builtin_amdgcn_wmma_f32_16x16x32_bf16(false, (AH).v, false, (BH).v, (short)0, (ACC).v, false, false); \
  (ACC).v = __builtin_amdgcn_wmma_f32_16x16x32_bf16(false, (AH).v, false, (BL).v, (short)0, (ACC).v, false, false); \
  (ACC).v = __builtin_amdgcn_wmma_f32_16x16x32_bf16(false, (AL).v, false, (BH).v, (short)0, (ACC).v, false, false)

// ---------------------------------------------------------------------------
// Kernel 0: pre-split W1 -> column-major pair-packed bf16 hi/lo (one buffer,
//           lo at dword offset W1HL), K zero-padded to 288.
// ---------------------------------------------------------------------------
__global__ void split_w1(const float* __restrict__ W1, unsigned int* __restrict__ hl) {
  const int idx = blockIdx.x * blockDim.x + threadIdx.x;
  if (idx >= HID * W1LD) return;
  const int col = idx / W1LD, p = idx % W1LD, k = 2 * p;
  float w0 = 0.f, w1v = 0.f;
  if (k < NN) { w0 = W1[(size_t)k * HID + col]; w1v = W1[(size_t)(k + 1) * HID + col]; }
  unsigned short h0, l0, h1, l1;
  split2(w0, h0, l0); split2(w1v, h1, l1);
  hl[(size_t)col * W1LD + p]        = pack2(h0, h1);
  hl[(size_t)col * W1LD + p + W1HL] = pack2(l0, l1);
}

// Kernel 0b: pre-split fc1_w -> column-major pair-packed bf16 hi/lo (one buffer)
__global__ void split_fc1(const float* __restrict__ w, unsigned int* __restrict__ hl) {
  const int idx = blockIdx.x * blockDim.x + threadIdx.x;
  if (idx >= 128 * FC1LD) return;
  const int col = idx / FC1LD, p = idx % FC1LD, k = 2 * p;
  const float w0  = w[(size_t)k * 128 + col];
  const float w1v = w[(size_t)(k + 1) * 128 + col];
  unsigned short h0, l0, h1, l1;
  split2(w0, h0, l0); split2(w1v, h1, l1);
  hl[(size_t)col * FC1LD + p]         = pack2(h0, h1);
  hl[(size_t)col * FC1LD + p + FC1HL] = pack2(l0, l1);
}

// ---------------------------------------------------------------------------
// Kernel 1: per-batch fused  support = x@W1 ; h = selu(adj@support + b1) ;
//           l1[b] = avgpool2x2(h).  One workgroup per graph, support in LDS
//           as pair-packed bf16 hi/lo (column-major, SUPLD pairs per column).
// ---------------------------------------------------------------------------
__global__ __launch_bounds__(512)
void gcn_fused(const float* __restrict__ x, const float* __restrict__ adj,
               const unsigned int* __restrict__ w1hl,
               const float* __restrict__ b1, float* __restrict__ l1) {
  extern __shared__ unsigned int ldsbuf[];
  unsigned int* sup_h = ldsbuf;                    // [HID][SUPLD]
  unsigned int* sup_l = ldsbuf + (size_t)HID * SUPLD;

  const int b    = blockIdx.x;
  const int lane = threadIdx.x & 31;
  const int wave = threadIdx.x >> 5;               // 16 waves
  const int half = lane >> 4;
  const int mloc = lane & 15;

  // zero K-padding pairs (rows 264..287 -> pairs 132..143)
  for (int i = threadIdx.x; i < HID * 12; i += blockDim.x) {
    const int col = i / 12, p = 132 + (i % 12);
    sup_h[(size_t)col * SUPLD + p] = 0u;
    sup_l[(size_t)col * SUPLD + p] = 0u;
  }
  __syncthreads();

  const float* xb  = x   + (size_t)b * NN * NN;
  const float* ajb = adj + (size_t)b * NN * NN;

  // ---------------- Phase 1: sup = x[b] @ W1 -------------------------------
  for (int tg = wave; tg < 34; tg += 16) {         // (row-tile, col-half) groups
    const int mt = tg >> 1;
    const int M0 = mt * 16;
    const int NB = (tg & 1) * 128;
    int mrow = M0 + mloc; if (mrow > NN - 1) mrow = NN - 1;
    const float* arow = xb + (size_t)mrow * NN;

    AccFrag acc[8];
    #pragma unroll
    for (int nt = 0; nt < 8; ++nt)
      #pragma unroll
      for (int j = 0; j < 8; ++j) acc[nt].f[j] = 0.f;

    const float*        arun = arow + 8 * half;
    const unsigned int* brun = w1hl + (size_t)(NB + mloc) * W1LD + 4 * half;

    #pragma unroll 2
    for (int ks = 0; ks < 8; ++ks) {               // clean K steps, no guards
      BF16Frag ah, al;
      load_split_A(arun, ah, al);
      #pragma unroll
      for (int nt = 0; nt < 8; ++nt) {
        const unsigned int* cb = brun + nt * (16 * W1LD);
        BF16Frag bh, bl;
        frag_from_u4(bh, *(const uint4*)cb,            *(const uint4*)(cb + 8));
        frag_from_u4(bl, *(const uint4*)(cb + W1HL),   *(const uint4*)(cb + W1HL + 8));
        WMMA3(acc[nt], ah, al, bh, bl);
      }
      arun += 32; brun += 16;
    }
    { // K tail (k = 256..287): only half==0, j<4 carries data
      BF16Frag ah, al;
      #pragma unroll
      for (int j = 0; j < 8; ++j) { ah.w[j] = 0u; al.w[j] = 0u; }
      if (half == 0) {
        #pragma unroll
        for (int j = 0; j < 4; ++j) {
          const float2 a = *(const float2*)(arow + 256 + 2 * j);
          split2(a.x, ah.u[2*j],   al.u[2*j]);
          split2(a.y, ah.u[2*j+1], al.u[2*j+1]);
        }
      }
      #pragma unroll
      for (int nt = 0; nt < 8; ++nt) {
        const unsigned int* cb = brun + nt * (16 * W1LD);
        BF16Frag bh, bl;
        frag_from_u4(bh, *(const uint4*)cb,            *(const uint4*)(cb + 8));
        frag_from_u4(bl, *(const uint4*)(cb + W1HL),   *(const uint4*)(cb + W1HL + 8));
        WMMA3(acc[nt], ah, al, bh, bl);
      }
    }
    // epilogue: split to bf16 hi/lo, pack row pairs, store column-major to LDS
    #pragma unroll
    for (int nt = 0; nt < 8; ++nt) {
      const int col = NB + nt * 16 + mloc;
      unsigned int* ch = sup_h + (size_t)col * SUPLD;
      unsigned int* cl = sup_l + (size_t)col * SUPLD;
      #pragma unroll
      for (int jj = 0; jj < 4; ++jj) {
        const int r0 = M0 + 2 * jj + 8 * half;
        if (r0 < NN) {
          unsigned short h0, l0, h1, l1;
          split2(acc[nt].f[2*jj],     h0, l0);
          split2(acc[nt].f[2*jj + 1], h1, l1);
          ch[r0 >> 1] = pack2(h0, h1);
          cl[r0 >> 1] = pack2(l0, l1);
        }
      }
    }
  }
  __syncthreads();

  // ---------------- Phase 2: h = adj[b] @ sup ; selu ; pool ----------------
  for (int tg = wave; tg < 34; tg += 16) {
    const int mt = tg >> 1;
    const int M0 = mt * 16;
    const int NB = (tg & 1) * 128;
    int mrow = M0 + mloc; if (mrow > NN - 1) mrow = NN - 1;
    const float* arow = ajb + (size_t)mrow * NN;

    AccFrag acc[8];
    #pragma unroll
    for (int nt = 0; nt < 8; ++nt)
      #pragma unroll
      for (int j = 0; j < 8; ++j) acc[nt].f[j] = 0.f;

    const float* arun = arow + 8 * half;
    const int    cb0  = (NB + mloc) * SUPLD + 4 * half;

    #pragma unroll 2
    for (int ks = 0; ks < 8; ++ks) {
      BF16Frag ah, al;
      load_split_A(arun, ah, al);
      #pragma unroll
      for (int nt = 0; nt < 8; ++nt) {
        const unsigned int* cbh = sup_h + cb0 + nt * (16 * SUPLD) + ks * 16;
        const unsigned int* cbl = sup_l + cb0 + nt * (16 * SUPLD) + ks * 16;
        BF16Frag bh, bl;
        frag_from_u4(bh, *(const uint4*)cbh, *(const uint4*)(cbh + 8));
        frag_from_u4(bl, *(const uint4*)cbl, *(const uint4*)(cbl + 8));
        WMMA3(acc[nt], ah, al, bh, bl);
      }
      arun += 32;
    }
    { // K tail
      BF16Frag ah, al;
      #pragma unroll
      for (int j = 0; j < 8; ++j) { ah.w[j] = 0u; al.w[j] = 0u; }
      if (half == 0) {
        #pragma unroll
        for (int j = 0; j < 4; ++j) {
          const float2 a = *(const float2*)(arow + 256 + 2 * j);
          split2(a.x, ah.u[2*j],   al.u[2*j]);
          split2(a.y, ah.u[2*j+1], al.u[2*j+1]);
        }
      }
      #pragma unroll
      for (int nt = 0; nt < 8; ++nt) {
        const unsigned int* cbh = sup_h + cb0 + nt * (16 * SUPLD) + 128;
        const unsigned int* cbl = sup_l + cb0 + nt * (16 * SUPLD) + 128;
        BF16Frag bh, bl;
        frag_from_u4(bh, *(const uint4*)cbh, *(const uint4*)(cbh + 8));
        frag_from_u4(bl, *(const uint4*)cbl, *(const uint4*)(cbl + 8));
        WMMA3(acc[nt], ah, al, bh, bl);
      }
    }
    // epilogue: bias + selu + 2x2 avg-pool, write l1[b]
    #pragma unroll
    for (int nt = 0; nt < 8; ++nt) {
      const int col = NB + nt * 16 + mloc;
      const float bias = b1[col];
      float v[8];
      #pragma unroll
      for (int j = 0; j < 8; ++j) v[j] = selu_f(acc[nt].f[j] + bias);
      #pragma unroll
      for (int jj = 0; jj < 4; ++jj) {
        float s = v[2*jj] + v[2*jj + 1];           // row pair (same lane)
        s += __shfl_xor(s, 1, 32);                 // column pair (lane^1)
        const int r0 = M0 + 2 * jj + 8 * half;
        if ((lane & 1) == 0 && r0 < NN) {
          const int pr = r0 >> 1, pc = col >> 1;
          l1[(size_t)b * FLAT + pr * 128 + pc] = s * 0.25f;
        }
      }
    }
  }
}

// ---------------------------------------------------------------------------
// Kernel 2: batch statistics per feature (mean, rstd)
// ---------------------------------------------------------------------------
__global__ void bn_stats(const float* __restrict__ l1, float* __restrict__ mu,
                         float* __restrict__ rstd) {
  const int f = blockIdx.x * blockDim.x + threadIdx.x;
  if (f >= FLAT) return;
  float s = 0.f, ss = 0.f;
  for (int i = 0; i < B_SZ; ++i) {
    const float v = l1[(size_t)i * FLAT + f];
    s += v; ss += v * v;
  }
  const float m   = s * (1.f / B_SZ);
  const float var = ss * (1.f / B_SZ) - m * m;
  mu[f]   = m;
  rstd[f] = rsqrtf(var + 1e-5f);
}

// ---------------------------------------------------------------------------
// Kernel 3: BN apply + bf16 hi/lo split (row-major: natural pair packing)
// ---------------------------------------------------------------------------
__global__ void bn_apply_split(const float* __restrict__ l1, const float* __restrict__ mu,
                               const float* __restrict__ rstd, const float* __restrict__ gamma,
                               const float* __restrict__ beta,
                               unsigned short* __restrict__ hi, unsigned short* __restrict__ lo) {
  const size_t idx = (size_t)blockIdx.x * blockDim.x + threadIdx.x;
  if (idx >= (size_t)B_SZ * FLAT) return;
  const int f = (int)(idx % FLAT);
  const float v = (l1[idx] - mu[f]) * rstd[f] * gamma[f] + beta[f];
  unsigned short h, l; split2(v, h, l);
  hi[idx] = h; lo[idx] = l;
}

// ---------------------------------------------------------------------------
// Kernel 4: l2 = selu(l1n @ fc1_w + b)  [1024,16896]x[16896,128]
//   64 row-tiles, one wave each with 8 column accumulators; K = 528*32 exact.
//   All B loads: immediate offsets from one running base pointer.
// ---------------------------------------------------------------------------
__global__ __launch_bounds__(256)
void fc1_gemm(const unsigned int* __restrict__ Ah, const unsigned int* __restrict__ Al,
              const unsigned int* __restrict__ Bhl,
              const float* __restrict__ bias, float* __restrict__ out) {
  const int lane = threadIdx.x & 31, wave = threadIdx.x >> 5;
  const int half = lane >> 4, nloc = lane & 15;
  const int mt = blockIdx.x * 8 + wave;            // 8 blocks * 8 waves = 64 tiles
  const int M0 = mt * 16;
  const int m  = M0 + nloc;

  const unsigned int* arun = Ah + (size_t)m * FC1LD + 4 * half;
  const unsigned int* alun = Al + (size_t)m * FC1LD + 4 * half;
  const unsigned int* brun = Bhl + (size_t)nloc * FC1LD + 4 * half;

  AccFrag acc[8];
  #pragma unroll
  for (int nt = 0; nt < 8; ++nt)
    #pragma unroll
    for (int j = 0; j < 8; ++j) acc[nt].f[j] = 0.f;

  #pragma unroll 2
  for (int ks = 0; ks < FLAT / 32; ++ks) {
    BF16Frag ah, al;
    frag_from_u4(ah, *(const uint4*)arun, *(const uint4*)(arun + 8));
    frag_from_u4(al, *(const uint4*)alun, *(const uint4*)(alun + 8));
    #pragma unroll
    for (int nt = 0; nt < 8; ++nt) {
      const unsigned int* cb = brun + nt * (16 * FC1LD);
      BF16Frag bh, bl;
      frag_from_u4(bh, *(const uint4*)cb,          *(const uint4*)(cb + 8));
      frag_from_u4(bl, *(const uint4*)(cb + FC1HL), *(const uint4*)(cb + FC1HL + 8));
      WMMA3(acc[nt], ah, al, bh, bl);
    }
    arun += 16; alun += 16; brun += 16;
  }
  #pragma unroll
  for (int nt = 0; nt < 8; ++nt) {
    const int col = nt * 16 + nloc;
    const float bv = bias[col];
    #pragma unroll
    for (int j = 0; j < 8; ++j) {
      const int r = M0 + j + 8 * half;
      out[(size_t)r * 128 + col] = selu_f(acc[nt].f[j] + bv);
    }
  }
}

// ---------------------------------------------------------------------------
// Kernel 5: out = sigmoid(l2 @ fc2_w + fc2_b)
// ---------------------------------------------------------------------------
__global__ void fc2_sigmoid(const float* __restrict__ l2, const float* __restrict__ w,
                            const float* __restrict__ bias, float* __restrict__ out) {
  const int r = blockIdx.x * blockDim.x + threadIdx.x;
  if (r >= B_SZ) return;
  float s = bias[0];
  #pragma unroll 8
  for (int j = 0; j < 128; ++j) s += l2[r * 128 + j] * w[j];
  out[r] = 1.f / (1.f + __expf(-s));
}

// ---------------------------------------------------------------------------
extern "C" void kernel_launch(void* const* d_in, const int* in_sizes, int n_in,
                              void* d_out, int out_size, void* d_ws, size_t ws_size,
                              hipStream_t stream) {
  (void)in_sizes; (void)n_in; (void)out_size; (void)ws_size;
  const float* x     = (const float*)d_in[0];
  const float* adj   = (const float*)d_in[1];
  const float* W1    = (const float*)d_in[2];
  const float* b1    = (const float*)d_in[3];
  const float* gamma = (const float*)d_in[4];
  const float* beta  = (const float*)d_in[5];
  const float* fc1w  = (const float*)d_in[6];
  const float* fc1b  = (const float*)d_in[7];
  const float* fc2w  = (const float*)d_in[8];
  const float* fc2b  = (const float*)d_in[9];
  float* out = (float*)d_out;

  char* ws = (char*)d_ws;
  size_t off = 0;
  auto alloc = [&](size_t bytes) -> void* {
    void* p = ws + off;
    off += (bytes + 255) & ~(size_t)255;
    return p;
  };
  float*          l1    = (float*)alloc((size_t)B_SZ * FLAT * sizeof(float));
  float*          mu    = (float*)alloc(FLAT * sizeof(float));
  float*          rstd  = (float*)alloc(FLAT * sizeof(float));
  unsigned short* l1h   = (unsigned short*)alloc((size_t)B_SZ * FLAT * 2);
  unsigned short* l1l   = (unsigned short*)alloc((size_t)B_SZ * FLAT * 2);
  unsigned int*   w1hl  = (unsigned int*)alloc((size_t)2 * HID * W1LD * 4);   // hi | lo
  unsigned int*   fc1hl = (unsigned int*)alloc((size_t)2 * 128 * FC1LD * 4);  // hi | lo
  float*          l2    = (float*)alloc((size_t)B_SZ * 128 * sizeof(float));

  split_w1<<<(HID * W1LD + 255) / 256, 256, 0, stream>>>(W1, w1hl);

  const size_t lds_bytes = (size_t)2 * HID * SUPLD * sizeof(unsigned int);  // 303,104 B
  gcn_fused<<<B_SZ, 512, lds_bytes, stream>>>(x, adj, w1hl, b1, l1);

  bn_stats<<<FLAT / 256, 256, 0, stream>>>(l1, mu, rstd);

  const size_t nelem = (size_t)B_SZ * FLAT;
  bn_apply_split<<<(unsigned)((nelem + 255) / 256), 256, 0, stream>>>(l1, mu, rstd, gamma, beta, l1h, l1l);

  split_fc1<<<(128 * FC1LD + 255) / 256, 256, 0, stream>>>(fc1w, fc1hl);

  fc1_gemm<<<8, 256, 0, stream>>>((const unsigned int*)l1h, (const unsigned int*)l1l,
                                  fc1hl, fc1b, l2);

  fc2_sigmoid<<<(B_SZ + 255) / 256, 256, 0, stream>>>(l2, fc2w, fc2b, out);
}
